// CausalSelfAttention_24034636988727
// MI455X (gfx1250) — matrix operand order repaired and checked by reading, not compiled, not run on hardware
//
#include <hip/hip_runtime.h>

// ---------------------------------------------------------------------------
// CDNA5 / gfx1250: cumsum-normalized linear attention.
// All matrix math on v_wmma_f32_16x16x32_bf16 (wave32 WMMA).
// bf16 conversion hoisted out of all GEMM inner loops; GEMM waves own a
// 32x64 tile so each B fragment feeds two WMMAs.
// ---------------------------------------------------------------------------

typedef __attribute__((ext_vector_type(16))) __bf16 v16bf;
typedef __attribute__((ext_vector_type(8)))  float  v8f;

#define T_SEQ   4096
#define C_DIM   768
#define N_HEAD  12
#define HD      64
#define EPS_DENOM 1e-6f
#define EPS_NORM  1e-12f

// native RNE conversion -> packed bf16 cvt ops
static __device__ __forceinline__ __bf16 f2bf(float f) { return (__bf16)f; }

static __device__ __forceinline__ v8f wmma_bf16(v16bf a, v16bf b, v8f c) {
  return __builtin_amdgcn_wmma_f32_16x16x32_bf16(
      /*neg_a=*/false, a, /*neg_b=*/false, b,
      /*c_mod=*/(short)0, c, /*reuse_a=*/false, /*reuse_b=*/false);
}

// ---- A fragment (16x32 bf16): lanes 0-15 m=lane K={0..7,16..23};
//      lanes 16-31 m=lane-16 K={8..15,24..31}; contiguous pair loads --------
static __device__ __forceinline__ v16bf load_a_bf(const __bf16* src,
                                                  int lda, int kbase, int lane) {
  const int m = lane & 15, kh = lane >> 4;
  v16bf a;
#pragma unroll
  for (int i = 0; i < 8; ++i) {
    const int ks = kbase + ((i < 4) ? (kh * 8 + 2 * i) : (16 + kh * 8 + 2 * (i - 4)));
    a[2 * i]     = src[(size_t)m * lda + ks];
    a[2 * i + 1] = src[(size_t)m * lda + ks + 1];
  }
  return a;
}

// ---- B fragment (32x16 bf16), source stored column-major (N x K row-major):
//      lanes 0-15 K={0..7,16..23}, lanes 16-31 K={8..15,24..31}; contiguous pair loads ------
static __device__ __forceinline__ v16bf load_bT_bf(const __bf16* src,
                                                   int ld, int kbase, int lane) {
  const int n = lane & 15, kh = lane >> 4;
  v16bf b;
#pragma unroll
  for (int i = 0; i < 8; ++i) {
    const size_t idx = (size_t)n * ld + kbase + ((i < 4) ? (kh * 8 + 2 * i) : (16 + kh * 8 + 2 * (i - 4)));
    b[2 * i]     = src[idx];
    b[2 * i + 1] = src[idx + 1];
  }
  return b;
}

// ---------------------------------------------------------------------------
// Elementwise f32 -> bf16 convert
// ---------------------------------------------------------------------------
__global__ void __launch_bounds__(256)
cvt_bf16(const float* __restrict__ in, unsigned short* __restrict__ out, int count) {
  const int i = blockIdx.x * 256 + threadIdx.x;
  if (i < count) ((__bf16*)out)[i] = f2bf(in[i]);
}

// f32 [K,N] row-major -> bf16 [N,K] (transposed) so GEMM B loads are
// contiguous per lane
__global__ void __launch_bounds__(256)
cvt_transpose_bf16(const float* __restrict__ in, unsigned short* __restrict__ out,
                   int K, int N) {
  const int i = blockIdx.x * 256 + threadIdx.x;
  if (i >= K * N) return;
  const int k = i / N, n = i - k * N;
  ((__bf16*)out)[(size_t)n * K + k] = f2bf(in[i]);
}

// ---------------------------------------------------------------------------
// GEMM: C[M,N] = A[M,K](bf16 row-major) @ Bt[N,K](bf16) + bias, fp32 out.
// One wave per block; 32x64 tile per wave: 2 A-frags x 4 B-frags = 8 WMMA
// per K-step, each B fragment feeds two WMMAs.
// ---------------------------------------------------------------------------
__global__ void __launch_bounds__(32)
gemm_bf_bias64(const unsigned short* __restrict__ A_,
               const unsigned short* __restrict__ Bt_,
               const float* __restrict__ bias, float* __restrict__ C,
               int N, int K) {
  const int lane = threadIdx.x;
  const int m0 = blockIdx.y * 32;
  const int n0 = blockIdx.x * 64;
  const __bf16* A0 = (const __bf16*)A_ + (size_t)m0 * K;
  const __bf16* A1 = A0 + (size_t)16 * K;
  const __bf16* Bt = (const __bf16*)Bt_ + (size_t)n0 * K;

  v8f acc[2][4];
#pragma unroll
  for (int r = 0; r < 2; ++r)
#pragma unroll
    for (int t = 0; t < 4; ++t)
#pragma unroll
      for (int i = 0; i < 8; ++i) acc[r][t][i] = 0.0f;

  for (int kb = 0; kb < K; kb += 32) {
    if (kb + 32 < K) {
      __builtin_prefetch(A0 + ((size_t)(lane & 15) * K + kb + 32), 0, 1);  // global_prefetch_b8
      __builtin_prefetch(A1 + ((size_t)(lane & 15) * K + kb + 32), 0, 1);
      __builtin_prefetch(Bt + ((size_t)(lane & 15) * K + kb + 32), 0, 1);
    }
    const v16bf a0 = load_a_bf(A0, K, kb, lane);
    const v16bf a1 = load_a_bf(A1, K, kb, lane);
#pragma unroll
    for (int t = 0; t < 4; ++t) {
      const v16bf b = load_bT_bf(Bt + (size_t)(t * 16) * K, K, kb, lane);
      acc[0][t] = wmma_bf16(a0, b, acc[0][t]);
      acc[1][t] = wmma_bf16(a1, b, acc[1][t]);
    }
  }

  const int n = lane & 15, mb = (lane >> 4) * 8;
#pragma unroll
  for (int r = 0; r < 2; ++r)
#pragma unroll
    for (int t = 0; t < 4; ++t)
#pragma unroll
      for (int v = 0; v < 8; ++v) {
        const int row = m0 + r * 16 + mb + v;
        const int col = n0 + t * 16 + n;
        C[(size_t)row * N + col] = acc[r][t][v] + bias[col];
      }
}

// ---------------------------------------------------------------------------
// Split QKV, L2-normalize q,k over hd=64, emit bf16:
//   q,k -> [H,T,64] row-major;  v -> [H,64,T] (transposed B-operand layout)
// One wave per (h,t) row; lane handles d=lane and d=lane+32.
// ---------------------------------------------------------------------------
__global__ void __launch_bounds__(128)
norm_split(const float* __restrict__ qkv, unsigned short* __restrict__ qus,
           unsigned short* __restrict__ kus, unsigned short* __restrict__ vus) {
  __bf16* qb = (__bf16*)qus;
  __bf16* kb = (__bf16*)kus;
  __bf16* vb = (__bf16*)vus;

  const int r    = blockIdx.x * 4 + (threadIdx.x >> 5);
  const int lane = threadIdx.x & 31;
  const int h = r >> 12;          // T = 4096
  const int t = r & (T_SEQ - 1);

  const float* row = qkv + (size_t)t * (3 * C_DIM);
  const int d0 = lane, d1 = lane + 32;

  const float q0 = row[h * HD + d0],              q1 = row[h * HD + d1];
  const float k0 = row[C_DIM + h * HD + d0],      k1 = row[C_DIM + h * HD + d1];
  const float v0 = row[2 * C_DIM + h * HD + d0],  v1 = row[2 * C_DIM + h * HD + d1];

  float sq = q0 * q0 + q1 * q1;
  float sk = k0 * k0 + k1 * k1;
#pragma unroll
  for (int off = 16; off > 0; off >>= 1) {
    sq += __shfl_xor(sq, off, 32);
    sk += __shfl_xor(sk, off, 32);
  }
  const float iq = 1.0f / fmaxf(sqrtf(sq), EPS_NORM);
  const float ik = 1.0f / fmaxf(sqrtf(sk), EPS_NORM);

  const size_t base = ((size_t)h * T_SEQ + t) * HD;
  qb[base + d0] = f2bf(q0 * iq);  qb[base + d1] = f2bf(q1 * iq);
  kb[base + d0] = f2bf(k0 * ik);  kb[base + d1] = f2bf(k1 * ik);

  const size_t vbase = (size_t)h * HD * T_SEQ;
  vb[vbase + (size_t)d0 * T_SEQ + t] = f2bf(v0);
  vb[vbase + (size_t)d1 * T_SEQ + t] = f2bf(v1);
}

// ---------------------------------------------------------------------------
// Streaming cumsum-normalized attention. Block = 1 wave = one (head, 16-query
// tile). Per 32-key block: num = q̂·k̂ᵀ (4 WMMA, K=64) -> LDS fp32 ->
// sequential cumsum + clamp + divide (per-row carry persists across blocks,
// att stored to LDS already in bf16) -> y += att·v (4 WMMA). y emitted bf16
// in [T,C] layout so the output projection consumes it directly.
// ---------------------------------------------------------------------------
__global__ void __launch_bounds__(32)
attn(const unsigned short* __restrict__ qus, const unsigned short* __restrict__ kus,
     const unsigned short* __restrict__ vus, unsigned short* __restrict__ yus) {
  const int lane = threadIdx.x;
  const int h  = blockIdx.y;
  const int q0 = blockIdx.x * 16;

  const __bf16* qh = (const __bf16*)qus + ((size_t)h * T_SEQ + q0) * HD;
  const __bf16* kh = (const __bf16*)kus + (size_t)h * T_SEQ * HD;
  const __bf16* vt = (const __bf16*)vus + (size_t)h * HD * T_SEQ;
  __bf16* yb = (__bf16*)yus;

  // q tile (16 x 64) resident as two A fragments
  const v16bf qa0 = load_a_bf(qh, HD, 0, lane);
  const v16bf qa1 = load_a_bf(qh, HD, 32, lane);

  v8f y[4];
#pragma unroll
  for (int t = 0; t < 4; ++t)
#pragma unroll
    for (int i = 0; i < 8; ++i) y[t][i] = 0.0f;

  __shared__ float  numt[16][34];   // fp32 num tile (padded)
  __shared__ __bf16 attt[16][36];   // bf16 att tile (padded)
  float carry = 0.0f;               // running cumsum; row = lane (lanes 0-15)

  const int n = lane & 15, mb = (lane >> 4) * 8;

  for (int j0 = 0; j0 < T_SEQ; j0 += 32) {
    // ---- num tile: 16 queries x 32 keys -------------------------------
#pragma unroll
    for (int t = 0; t < 2; ++t) {
      const __bf16* ksrc = kh + (size_t)(j0 + t * 16) * HD;  // rows = keys
      const v16bf b0 = load_bT_bf(ksrc, HD, 0, lane);
      const v16bf b1 = load_bT_bf(ksrc, HD, 32, lane);
      v8f c;
#pragma unroll
      for (int i = 0; i < 8; ++i) c[i] = 0.0f;
      c = wmma_bf16(qa0, b0, c);
      c = wmma_bf16(qa1, b1, c);
#pragma unroll
      for (int v = 0; v < 8; ++v) numt[mb + v][t * 16 + n] = c[v];
    }
    __syncthreads();

    // ---- sequential cumsum + clamp + divide; emit bf16 att ------------
    if (lane < 16) {
      float c = carry;
#pragma unroll 4
      for (int j = 0; j < 32; ++j) {
        const float x = numt[lane][j];
        c += x;
        attt[lane][j] = f2bf(x / fmaxf(c, EPS_DENOM));
      }
      carry = c;
    }
    __syncthreads();

    // ---- y += att (16x32) @ v (32x64) ---------------------------------
    const v16bf aa = load_a_bf(&attt[0][0], 36, 0, lane);
#pragma unroll
    for (int t = 0; t < 4; ++t) {
      const v16bf vb = load_bT_bf(vt + (size_t)(t * 16) * T_SEQ, T_SEQ, j0, lane);
      y[t] = wmma_bf16(aa, vb, y[t]);
    }
    __syncthreads();
  }

  // ---- store y bf16 into [T, C]: col = h*64 + d -------------------------
#pragma unroll
  for (int t = 0; t < 4; ++t)
#pragma unroll
    for (int v = 0; v < 8; ++v) {
      const int row = q0 + mb + v;
      const int col = h * HD + t * 16 + n;
      yb[(size_t)row * C_DIM + col] = f2bf(y[t][v]);
    }
}

// ---------------------------------------------------------------------------
extern "C" void kernel_launch(void* const* d_in, const int* in_sizes, int n_in,
                              void* d_out, int out_size, void* d_ws, size_t ws_size,
                              hipStream_t stream) {
  const float* x      = (const float*)d_in[0];   // [T, C]
  const float* w_attn = (const float*)d_in[1];   // [C, 3C]
  const float* b_attn = (const float*)d_in[2];   // [3C]
  const float* w_proj = (const float*)d_in[3];   // [C, C]
  const float* b_proj = (const float*)d_in[4];   // [C]
  float* out = (float*)d_out;                    // [T, C]

  // ---- workspace layout (bytes) ----
  const size_t QKV_B = (size_t)T_SEQ * 3 * C_DIM * sizeof(float);        // 37,748,736
  const size_t HEADE = (size_t)N_HEAD * T_SEQ * HD;                      //  3,145,728
  const size_t HBF_B = HEADE * 2;                                        //  6,291,456
  const size_t XBF_B = (size_t)T_SEQ * C_DIM * 2;                        //  6,291,456
  const size_t WAT_B = (size_t)C_DIM * 3 * C_DIM * 2;                    //  3,538,944
  const size_t WPT_B = (size_t)C_DIM * C_DIM * 2;                        //  1,179,648
  const size_t YBF_B = (size_t)T_SEQ * C_DIM * 2;                        //  6,291,456
  const size_t NEED  = QKV_B + 3 * HBF_B + XBF_B + WAT_B + WPT_B + YBF_B;
  if (ws_size < NEED) return;

  char* ws = (char*)d_ws;
  float*          qkv = (float*)ws;
  unsigned short* qbf = (unsigned short*)(ws + QKV_B);
  unsigned short* kbf = qbf + HEADE;
  unsigned short* vbf = kbf + HEADE;
  unsigned short* xbf = (unsigned short*)(ws + QKV_B + 3 * HBF_B);
  unsigned short* wat = (unsigned short*)((char*)xbf + XBF_B);
  unsigned short* wpt = (unsigned short*)((char*)wat + WAT_B);
  unsigned short* ybf = (unsigned short*)((char*)wpt + WPT_B);

  // 0) one-shot bf16 conversions (weights transposed to [N,K])
  {
    const int nx = T_SEQ * C_DIM;
    cvt_bf16<<<(nx + 255) / 256, 256, 0, stream>>>(x, xbf, nx);
    const int nwa = C_DIM * 3 * C_DIM;
    cvt_transpose_bf16<<<(nwa + 255) / 256, 256, 0, stream>>>(w_attn, wat, C_DIM, 3 * C_DIM);
    const int nwp = C_DIM * C_DIM;
    cvt_transpose_bf16<<<(nwp + 255) / 256, 256, 0, stream>>>(w_proj, wpt, C_DIM, C_DIM);
  }

  // 1) QKV projection: [4096,768] x [768,2304] + bias -> fp32
  gemm_bf_bias64<<<dim3(3 * C_DIM / 64, T_SEQ / 32), 32, 0, stream>>>(
      xbf, wat, b_attn, qkv, 3 * C_DIM, C_DIM);

  // 2) split + L2 normalize + transpose v (bf16 out)
  norm_split<<<(N_HEAD * T_SEQ) / 4, 128, 0, stream>>>(qkv, qbf, kbf, vbf);

  // 3) streaming cumsum-normalized attention (bf16 y out)
  attn<<<dim3(T_SEQ / 16, N_HEAD), 32, 0, stream>>>(qbf, kbf, vbf, ybf);

  // 4) output projection: [4096,768] x [768,768] + bias -> fp32 out
  gemm_bf_bias64<<<dim3(C_DIM / 64, T_SEQ / 32), 32, 0, stream>>>(
      ybf, wpt, b_proj, out, C_DIM, C_DIM);
}
